// SlideAttention_6691559047251
// MI455X (gfx1250) — matrix operand order repaired and checked by reading, not compiled, not run on hardware
//
#include <hip/hip_runtime.h>
#include <hip/hip_bf16.h>

// ---------------------------------------------------------------------------
// SlideAttention for MI455X (gfx1250), wave32 + WMMA bf16, vectorized loads.
//
//   1. cvt kernels : inputs -> bf16; qkv_w/proj_w -> transposed bf16 [N][K];
//                    deform kernel -> two padded transposed bf16 copies
//                    (K-dim innermost, 64-padded) so every WMMA B operand
//                    half is a single aligned global_load_b128.
//   2. cpb_kernel  : SwinV2 continuous-position-bias MLP -> [8][49]
//   3. qkv_gemm    : WMMA GEMM [8192x256]@[256x768]; q->f32, k/v->bf16
//   4. attn_kernel : fused deform-conv + l2norm + softmax + output per
//                    (16-pixel tile, head) block. Deformed K'/V' (411 MB each
//                    if materialized) never touch HBM: K' streams into
//                    register partials of (n2, q.k'); V' is folded after
//                    softmax as attn @ DK^T then a window-weighted reduce.
//   5. proj_gemm   : WMMA GEMM [8192x256]@[256x256] + bias -> d_out (f32)
// ---------------------------------------------------------------------------

typedef __attribute__((ext_vector_type(16))) __bf16        v16bf;
typedef __attribute__((ext_vector_type(8)))  float         v8f;
typedef __attribute__((ext_vector_type(8)))  unsigned int  v8u;

#define BATCH 2
#define HH 64
#define WW 64
#define CC 256
#define HEADS 8
#define UNITS 32
#define WS 7
#define WS2 49
#define NPIX (BATCH*HH*WW)      // 8192
#define WSP 24                  // padded window width (22 used)
#define EPS 1.55e-05f

static __device__ __forceinline__ unsigned short f2bf(float f) {
  unsigned u = __builtin_bit_cast(unsigned, f);
  unsigned r = (u + 0x7FFFu + ((u >> 16) & 1u)) >> 16;   // round-nearest-even
  return (unsigned short)r;
}
static __device__ __forceinline__ __bf16 bf_bits(unsigned short b) {
  return __builtin_bit_cast(__bf16, b);
}
static __device__ __forceinline__ float bf2f(unsigned short b) {
  return __builtin_bit_cast(float, (unsigned)b << 16);
}
static __device__ __forceinline__ v8f wmma_bf16(v16bf a, v16bf b, v8f c) {
  return __builtin_amdgcn_wmma_f32_16x16x32_bf16(false, a, false, b,
                                                 (short)0, c, false, false);
}
static __device__ __forceinline__ v16bf mk16(uint4 lo, uint4 hi) {
  union { uint4 q[2]; v16bf v; } u;
  u.q[0] = lo; u.q[1] = hi;
  return u.v;
}
static __device__ __forceinline__ v16bf ld16bf(const unsigned short* p) {
  return __builtin_bit_cast(v16bf, *(const v8u*)p);      // 32B aligned load
}

// --------------------------- elementwise convert ---------------------------
__global__ __launch_bounds__(256) void cvt_bf16(const float* __restrict__ s,
                                                unsigned short* __restrict__ d,
                                                int n) {
  int i = blockIdx.x * 256 + threadIdx.x;
  if (i < n) d[i] = f2bf(s[i]);
}

// src [K][N] f32 -> dst [N][K] bf16
__global__ __launch_bounds__(256) void cvt_bf16_t(const float* __restrict__ s,
                                                  unsigned short* __restrict__ d,
                                                  int K, int N) {
  int i = blockIdx.x * 256 + threadIdx.x;
  if (i < K * N) {
    int k = i / N, n = i - k * N;
    d[n * K + k] = f2bf(s[i]);
  }
}

// deform kernel f32 [p=49][c=256][i=49] ->
//   dkA bf16 [c][i(64)][p(64)]  (phase-1 B: K=p innermost)
//   dkB bf16 [c][p(64)][i(64)]  (phase-3 B: K=i innermost)
__global__ __launch_bounds__(256) void dk_prep(const float* __restrict__ s,
                                               unsigned short* __restrict__ dA,
                                               unsigned short* __restrict__ dB) {
  int idx = blockIdx.x * 256 + threadIdx.x;        // 256*64*64
  if (idx >= CC * 64 * 64) return;
  int c = idx >> 12;
  int a = (idx >> 6) & 63;   // outer padded dim
  int b = idx & 63;          // inner padded dim
  unsigned short va = 0, vb = 0;
  if (a < WS2 && b < WS2) {
    va = f2bf(s[(size_t)b * CC * WS2 + (size_t)c * WS2 + a]);  // [i=a][p=b]
    vb = f2bf(s[(size_t)a * CC * WS2 + (size_t)c * WS2 + b]);  // [p=a][i=b]
  }
  dA[idx] = va;
  dB[idx] = vb;
}

// --------------------------- CPB bias MLP ----------------------------------
__global__ __launch_bounds__(256) void cpb_kernel(const float* __restrict__ w1,
                                                  const float* __restrict__ b1,
                                                  const float* __restrict__ w2,
                                                  float* __restrict__ out) {
  const float inv_log8 = 0.48089834696298783f;  // 1/ln(8)
  for (int idx = threadIdx.x; idx < WS2 * HEADS; idx += 256) {
    int tap = idx >> 3, head = idx & 7;
    float ry = (float)(tap / WS) - 3.0f;
    float rx = (float)(tap % WS) - 3.0f;
    float cy = ry / 6.0f * 8.0f, cx = rx / 6.0f * 8.0f;
    float c0 = copysignf(log1pf(fabsf(cy)) * inv_log8, cy);
    float c1 = copysignf(log1pf(fabsf(cx)) * inv_log8, cx);
    float s = 0.f;
    for (int j = 0; j < 512; ++j) {
      float hd = fmaxf(c0 * w1[j] + c1 * w1[512 + j] + b1[j], 0.f);
      s += hd * w2[j * HEADS + head];
    }
    out[head * WS2 + tap] = 16.f / (1.f + __expf(-s));
  }
}

// --------------------------- QKV GEMM (WMMA) -------------------------------
// M=8192 K=256 N=768; A bf16 row-major, B bf16 pre-transposed [N][K].
__global__ __launch_bounds__(256) void qkv_gemm(
    const unsigned short* __restrict__ xb, const unsigned short* __restrict__ wt,
    const float* __restrict__ qb, const float* __restrict__ vb,
    float* __restrict__ qout, unsigned short* __restrict__ kout,
    unsigned short* __restrict__ vout) {
  int wave = threadIdx.x >> 5, lane = threadIdx.x & 31;
  int row0 = blockIdx.x * 16;
  int ntile = blockIdx.y * 8 + wave;               // 0..47
  int m = lane & 15, hi = lane >> 4;
  int kb_a = hi * 16, kb_b = hi * 16;
  int col = ntile * 16 + (lane & 15);
  const unsigned short* arow = xb + (size_t)(row0 + m) * CC;
  const unsigned short* brow = wt + (size_t)col * CC;
  v8f acc = {};
  for (int kk = 0; kk < 256; kk += 32) {
    uint4 alo = *(const uint4*)(arow + kk + kb_a);
    uint4 ahi = *(const uint4*)(arow + kk + kb_a + 8);
    v16bf a = mk16(alo, ahi);
    v16bf b = ld16bf(brow + kk + kb_b);
    acc = wmma_bf16(a, b, acc);
  }
#pragma unroll
  for (int r = 0; r < 8; ++r) {
    int row = row0 + r + 8 * hi;                   // C 16x16 f32 layout
    float v = acc[r];
    if (col < 256)        qout[row * 256 + col] = v + qb[col];
    else if (col < 512)   kout[row * 256 + (col - 256)] = f2bf(v);
    else                  vout[row * 256 + (col - 512)] = f2bf(v + vb[col - 512]);
  }
}

// --------------------------- fused attention -------------------------------
// One block = (16-pixel row tile, head). blockDim=256 (8 waves).
__global__ __launch_bounds__(256) void attn_kernel(
    const unsigned short* __restrict__ kbuf,
    const unsigned short* __restrict__ vbuf,
    const float* __restrict__ qbuf,
    const unsigned short* __restrict__ dkA,   // [c][i64][p64] bf16
    const unsigned short* __restrict__ dkB,   // [c][p64][i64] bf16
    const float* __restrict__ scale,
    const float* __restrict__ cpb,
    unsigned short* __restrict__ aout) {
  __shared__ unsigned short kwin[WS][WSP][UNITS];
  __shared__ unsigned short vwin[WS][WSP][UNITS];
  __shared__ float qn[16][UNITS];
  __shared__ float qsc[16];
  __shared__ float n2a[16][52];
  __shared__ float da[16][52];
  __shared__ unsigned short attnb[16][64];
  __shared__ float outacc[16][UNITS];

  int tid = threadIdx.x;
  int bid = blockIdx.x;
  int head = bid & 7;
  int t = bid >> 3;                 // 0..511 pixel tiles
  int wt = t & 3;
  int rowhb = t >> 2;               // b*64+h
  int h = rowhb & 63;
  int w0 = wt * 16;
  int cbase = head * UNITS;

  // stage K/V window slabs (zero padded), 8 channels per b128 transaction
  for (int idx = tid; idx < WS * WSP * 4; idx += 256) {
    int cg = (idx & 3) * 8;
    int col = (idx >> 2) % WSP;
    int dy = idx / (WSP * 4);
    uint4 kv = {0, 0, 0, 0}, vv = {0, 0, 0, 0};
    int gy = h + dy - 3, gx = w0 + col - 3;
    if (col < 22 && gy >= 0 && gy < HH && gx >= 0 && gx < WW) {
      size_t gi = ((size_t)((rowhb >> 6) * HH + gy) * WW + gx) * CC + cbase + cg;
      kv = *(const uint4*)(kbuf + gi);
      vv = *(const uint4*)(vbuf + gi);
    }
    *(uint4*)&kwin[dy][col][cg] = kv;
    *(uint4*)&vwin[dy][col][cg] = vv;
  }
  for (int idx = tid; idx < 16 * UNITS; idx += 256) {
    int px = idx >> 5, c = idx & 31;
    qn[px][c] = qbuf[(size_t)(rowhb * WW + w0 + px) * CC + cbase + c];
    outacc[px][c] = 0.f;
  }
  for (int idx = tid; idx < 16 * 52; idx += 256) {
    n2a[idx / 52][idx % 52] = 0.f;
    da[idx / 52][idx % 52] = 0.f;
  }
  for (int idx = tid; idx < 16 * 32; idx += 256)
    *(unsigned*)&attnb[idx >> 5][(idx & 31) * 2] = 0u;
  __syncthreads();

  if (tid < 16) {
    float ss = 0.f;
    for (int c = 0; c < UNITS; ++c) ss += qn[tid][c] * qn[tid][c];
    qsc[tid] = rsqrtf(fmaxf(ss, EPS)) * __expf(scale[head]);
  }
  __syncthreads();
  for (int idx = tid; idx < 16 * UNITS; idx += 256)
    qn[idx >> 5][idx & 31] *= qsc[idx >> 5];
  __syncthreads();

  int wave = tid >> 5, lane = tid & 31;
  int m = lane & 15, hi = lane >> 4;
  int kb_a = hi * 16, kb_b = hi * 16;
  int n16 = lane & 15;

  // Phase 1: K' = window @ DK per channel via WMMA; register partials for
  // n2 = sum_c K'^2 and d = sum_c qn_c * K'  (positions channel-invariant).
  {
    float n2r[32], dr[32];
#pragma unroll
    for (int z = 0; z < 32; ++z) { n2r[z] = 0.f; dr[z] = 0.f; }
    for (int c = wave; c < UNITS; c += 8) {
      v16bf a0, a1;
#pragma unroll
      for (int e = 0; e < 16; ++e) {
        int kl = kb_a + e;
        int p1 = 32 + kl;
        a0[e] = bf_bits(kwin[kl / WS][m + kl % WS][c]);
        a1[e] = (p1 < WS2) ? bf_bits(kwin[p1 / WS][m + p1 % WS][c])
                           : (__bf16)0.0f;
      }
      float qv[8];
#pragma unroll
      for (int r = 0; r < 8; ++r) qv[r] = qn[r + 8 * hi][c];
      const unsigned short* base = dkA + ((size_t)(cbase + c) << 12);
#pragma unroll
      for (int nt = 0; nt < 4; ++nt) {
        int i = nt * 16 + n16;
        v16bf b0 = ld16bf(base + ((size_t)i << 6) + kb_b);
        v16bf b1 = ld16bf(base + ((size_t)i << 6) + 32 + kb_b);
        v8f acc = {};
        acc = wmma_bf16(a0, b0, acc);
        acc = wmma_bf16(a1, b1, acc);
#pragma unroll
        for (int r = 0; r < 8; ++r) {
          float kv = acc[r];
          n2r[nt * 8 + r] += kv * kv;
          dr[nt * 8 + r] += qv[r] * kv;
        }
      }
    }
#pragma unroll
    for (int nt = 0; nt < 4; ++nt) {
      int i = nt * 16 + n16;
      if (i < WS2) {
#pragma unroll
        for (int r = 0; r < 8; ++r) {
          int mm = r + 8 * hi;
          atomicAdd(&n2a[mm][i], n2r[nt * 8 + r]);
          atomicAdd(&da[mm][i], dr[nt * 8 + r]);
        }
      }
    }
  }
  __syncthreads();

  // Phase 2: logits = scaled q.k'_norm + border mask + CPB bias; softmax.
  if (tid < 16) {
    int px = tid;
    float lg[WS2];
    float mx = -1e30f;
    for (int i = 0; i < WS2; ++i) {
      int gy = h + i / WS - 3, gx = w0 + px + i % WS - 3;
      float msk = (gy >= 0 && gy < HH && gx >= 0 && gx < WW) ? 0.f : -100.f;
      float l = da[px][i] * rsqrtf(fmaxf(n2a[px][i], EPS)) + msk +
                cpb[head * WS2 + i];
      lg[i] = l;
      mx = fmaxf(mx, l);
    }
    float s = 0.f;
    for (int i = 0; i < WS2; ++i) { lg[i] = __expf(lg[i] - mx); s += lg[i]; }
    float inv = 1.f / s;
    for (int i = 0; i < WS2; ++i) attnb[px][i] = f2bf(lg[i] * inv);
  }
  __syncthreads();

  // Phase 3: out_c = sum_p v_win[p,c] * (attn @ DK^T)[p,c]  (WMMA per chan)
  v16bf at0, at1;
#pragma unroll
  for (int e = 0; e < 16; ++e) {
    int kl = kb_a + e;
    at0[e] = bf_bits(attnb[m][kl]);
    at1[e] = bf_bits(attnb[m][32 + kl]);
  }
  for (int c = wave; c < UNITS; c += 8) {
    float outr[8];
#pragma unroll
    for (int r = 0; r < 8; ++r) outr[r] = 0.f;
    const unsigned short* base = dkB + ((size_t)(cbase + c) << 12);
#pragma unroll
    for (int nt = 0; nt < 4; ++nt) {
      int p = nt * 16 + n16;
      v16bf b0 = ld16bf(base + ((size_t)p << 6) + kb_b);
      v16bf b1 = ld16bf(base + ((size_t)p << 6) + 32 + kb_b);
      v8f acc = {};
      acc = wmma_bf16(at0, b0, acc);
      acc = wmma_bf16(at1, b1, acc);
      if (p < WS2) {
        int dy = p / WS, dx = p % WS;
#pragma unroll
        for (int r = 0; r < 8; ++r)
          outr[r] += acc[r] * bf2f(vwin[dy][r + 8 * hi + dx][c]);
      }
    }
#pragma unroll
    for (int r = 0; r < 8; ++r) atomicAdd(&outacc[r + 8 * hi][c], outr[r]);
  }
  __syncthreads();

  // packed bf16x2 output write
  for (int idx = tid; idx < 16 * 16; idx += 256) {
    int px = idx >> 4, c2 = (idx & 15) * 2;
    unsigned lo = f2bf(outacc[px][c2]);
    unsigned hh = f2bf(outacc[px][c2 + 1]);
    *(unsigned*)(aout + (size_t)(rowhb * WW + w0 + px) * CC + cbase + c2) =
        lo | (hh << 16);
  }
}

// --------------------------- projection GEMM (WMMA) ------------------------
__global__ __launch_bounds__(256) void proj_gemm(
    const unsigned short* __restrict__ a, const unsigned short* __restrict__ wt,
    const float* __restrict__ pb, float* __restrict__ out) {
  int wave = threadIdx.x >> 5, lane = threadIdx.x & 31;
  int row0 = blockIdx.x * 16;
  int ntile = blockIdx.y * 8 + wave;                // 0..15
  int m = lane & 15, hi = lane >> 4;
  int kb_a = hi * 16, kb_b = hi * 16;
  int col = ntile * 16 + (lane & 15);
  const unsigned short* arow = a + (size_t)(row0 + m) * CC;
  const unsigned short* brow = wt + (size_t)col * CC;
  v8f acc = {};
  for (int kk = 0; kk < 256; kk += 32) {
    uint4 alo = *(const uint4*)(arow + kk + kb_a);
    uint4 ahi = *(const uint4*)(arow + kk + kb_a + 8);
    v16bf av = mk16(alo, ahi);
    v16bf bv = ld16bf(brow + kk + kb_b);
    acc = wmma_bf16(av, bv, acc);
  }
#pragma unroll
  for (int r = 0; r < 8; ++r) {
    int row = row0 + r + 8 * hi;
    out[row * 256 + col] = acc[r] + pb[col];
  }
}

// ---------------------------------------------------------------------------
extern "C" void kernel_launch(void* const* d_in, const int* in_sizes, int n_in,
                              void* d_out, int out_size, void* d_ws,
                              size_t ws_size, hipStream_t stream) {
  (void)in_sizes; (void)n_in; (void)out_size; (void)ws_size;
  const float* inputs = (const float*)d_in[0];
  const float* qkv_w  = (const float*)d_in[1];
  const float* q_bias = (const float*)d_in[2];
  const float* v_bias = (const float*)d_in[3];
  const float* dk     = (const float*)d_in[4];
  const float* scale  = (const float*)d_in[5];
  const float* cpb_w1 = (const float*)d_in[6];
  const float* cpb_b1 = (const float*)d_in[7];
  const float* cpb_w2 = (const float*)d_in[8];
  const float* proj_w = (const float*)d_in[9];
  const float* proj_b = (const float*)d_in[10];

  char* ws = (char*)d_ws;
  float*          qbuf   = (float*)(ws + 0);                  // 8192*256 f32
  unsigned short* kbuf   = (unsigned short*)(ws + 8388608);   // 8192*256 bf16
  unsigned short* vbuf   = (unsigned short*)(ws + 12582912);  // 8192*256 bf16
  unsigned short* abuf   = (unsigned short*)(ws + 16777216);  // 8192*256 bf16
  unsigned short* xbf    = (unsigned short*)(ws + 20971520);  // 8192*256 bf16
  unsigned short* wqkvT  = (unsigned short*)(ws + 25165824);  // 768*256 bf16
  unsigned short* wprojT = (unsigned short*)(ws + 25559040);  // 256*256 bf16
  unsigned short* dkA    = (unsigned short*)(ws + 25690112);  // 256*64*64 bf16
  unsigned short* dkB    = (unsigned short*)(ws + 27787264);  // 256*64*64 bf16
  float*          cpb    = (float*)(ws + 29884416);           // 8*49 f32

  cvt_bf16<<<(NPIX * CC + 255) / 256, 256, 0, stream>>>(inputs, xbf, NPIX * CC);
  cvt_bf16_t<<<(256 * 768 + 255) / 256, 256, 0, stream>>>(qkv_w, wqkvT, 256,
                                                          768);
  cvt_bf16_t<<<(256 * 256 + 255) / 256, 256, 0, stream>>>(proj_w, wprojT, 256,
                                                          256);
  dk_prep<<<(CC * 64 * 64 + 255) / 256, 256, 0, stream>>>(dk, dkA, dkB);
  cpb_kernel<<<1, 256, 0, stream>>>(cpb_w1, cpb_b1, cpb_w2, cpb);

  qkv_gemm<<<dim3(NPIX / 16, 6), 256, 0, stream>>>(xbf, wqkvT, q_bias, v_bias,
                                                   qbuf, kbuf, vbuf);
  attn_kernel<<<(NPIX / 16) * HEADS, 256, 0, stream>>>(kbuf, vbuf, qbuf, dkA,
                                                       dkB, scale, cpb, abuf);
  proj_gemm<<<dim3(NPIX / 16, 2), 256, 0, stream>>>(abuf, wprojT, proj_b,
                                                    (float*)d_out);
}
